// GCN_cora_21122649162596
// MI455X (gfx1250) — matrix operand order repaired and checked by reading, not compiled, not run on hardware
//
#include <hip/hip_runtime.h>
#include <hip/hip_bf16.h>

#define NFEAT 256
#define BN_EPS 1e-5f

typedef __attribute__((ext_vector_type(16))) __bf16        v16bf;
typedef __attribute__((ext_vector_type(8)))  float         v8f;
typedef __attribute__((ext_vector_type(4)))  unsigned int  v4u;

union FragBF { v16bf bf; v4u u[2]; };

__device__ __forceinline__ unsigned short f2bf(float f) {
    unsigned int u = __float_as_uint(f);
    u = (u + 0x7FFFu + ((u >> 16) & 1u)) >> 16;   // round-to-nearest-even
    return (unsigned short)u;
}

__device__ __forceinline__ v8f wmma_bf16(const FragBF& a, const FragBF& b, v8f c) {
    return __builtin_amdgcn_wmma_f32_16x16x32_bf16(
        false, a.bf, false, b.bf, (short)0, c, false, false);
}

// ---------------- utility kernels ----------------

__global__ void zero_f32(float* __restrict__ p, long n) {
    long i = (long)blockIdx.x * blockDim.x + threadIdx.x;
    long stride = (long)gridDim.x * blockDim.x;
    for (; i < n; i += stride) p[i] = 0.0f;
}

__global__ void cvt_f32_bf16(const float* __restrict__ x, unsigned short* __restrict__ y, long n) {
    long i = (long)blockIdx.x * blockDim.x + threadIdx.x;
    long stride = (long)gridDim.x * blockDim.x;
    for (; i < n; i += stride) y[i] = f2bf(x[i]);
}

// Pack W (row-major [K, realN] f32) into per-lane WMMA B-fragment order (bf16):
// dword index = ((kt*NT + nt)*32 + lane)*8 + v ; holds W[k,n],W[k+1,n]
// with k = kt*32 + (v>>2)*16 + (lane>>4)*8 + 2(v&3), n = nt*16 + (lane&15). Zero-pad n >= realN.
__global__ void pack_w_bf16(const float* __restrict__ W, unsigned int* __restrict__ Bp,
                            int KT, int NT, int realN) {
    int total = KT * NT * 256;
    for (int idx = blockIdx.x * blockDim.x + threadIdx.x; idx < total;
         idx += gridDim.x * blockDim.x) {
        int v    = idx & 7;
        int lane = (idx >> 3) & 31;
        int tile = idx >> 8;
        int nt = tile % NT;
        int kt = tile / NT;
        int k = kt * 32 + (v >> 2) * 16 + (lane >> 4) * 8 + 2 * (v & 3);
        int n = nt * 16 + (lane & 15);
        unsigned short b0 = 0, b1 = 0;
        if (n < realN) {
            b0 = f2bf(W[(size_t)k * realN + n]);
            b1 = f2bf(W[(size_t)(k + 1) * realN + n]);
        }
        Bp[idx] = (unsigned int)b0 | ((unsigned int)b1 << 16);
    }
}

// ---------------- 2x2 register-blocked WMMA GEMM (layers 1-2) ----------------
// One wave computes a 32x32 C tile: 4 accumulators, 4 WMMAs per K-step of 32.
// Requires M % 32 == 0 (N_NODES = 20000 = 625*32).
__global__ __launch_bounds__(256) void gemm_bf16_wmma_2x2(
    const unsigned short* __restrict__ A, const unsigned int* __restrict__ Bp,
    float* __restrict__ C, int Mtiles2, int NT, int ldc)
{
    const int wave = threadIdx.x >> 5;
    const int lane = threadIdx.x & 31;
    const int mt2 = blockIdx.x * 8 + wave;    // wave-uniform guard: EXEC all-1s
    const int nt2 = blockIdx.y;
    if (mt2 >= Mtiles2) return;

    const int g = lane >> 4;
    const int mrow = mt2 * 32 + (lane & 15);
    const unsigned short* arow0 = A + (size_t)mrow * NFEAT + g * 8;
    const unsigned short* arow1 = arow0 + (size_t)16 * NFEAT;
    const v4u* bb0 = reinterpret_cast<const v4u*>(Bp) + ((size_t)(nt2 * 2) * 32 + lane) * 2;
    const v4u* bb1 = bb0 + 64;                // adjacent 16-col tile
    const size_t bstep = (size_t)NT * 64;     // v4u per K-tile of 32

    v8f acc00 = {}, acc01 = {}, acc10 = {}, acc11 = {};
    #pragma unroll
    for (int kt = 0; kt < 8; ++kt) {
        FragBF a0, a1, b0, b1;
        const v4u* ap0 = reinterpret_cast<const v4u*>(arow0 + kt * 32);
        const v4u* ap1 = reinterpret_cast<const v4u*>(arow1 + kt * 32);
        a0.u[0] = ap0[0]; a0.u[1] = ap0[2];   // K = kt*32+g*8 .. / +16 ..
        a1.u[0] = ap1[0]; a1.u[1] = ap1[2];
        const v4u* bp0 = bb0 + (size_t)kt * bstep;
        const v4u* bp1 = bb1 + (size_t)kt * bstep;
        b0.u[0] = bp0[0]; b0.u[1] = bp0[1];
        b1.u[0] = bp1[0]; b1.u[1] = bp1[1];
        acc00 = wmma_bf16(a0, b0, acc00);
        acc01 = wmma_bf16(a0, b1, acc01);
        acc10 = wmma_bf16(a1, b0, acc10);
        acc11 = wmma_bf16(a1, b1, acc11);
    }

    const int ncol = nt2 * 32 + (lane & 15);
    float* c0 = C + (size_t)(mt2 * 32 + g * 8) * ldc + ncol;
    float* c1 = c0 + (size_t)16 * ldc;
    #pragma unroll
    for (int r = 0; r < 8; ++r) {
        c0[(size_t)r * ldc]      = acc00[r];
        c0[(size_t)r * ldc + 16] = acc01[r];
        c1[(size_t)r * ldc]      = acc10[r];
        c1[(size_t)r * ldc + 16] = acc11[r];
    }
}

// ---------------- 16x16-per-wave WMMA GEMM (layer 3, 48 padded cols) --------
__global__ __launch_bounds__(256) void gemm_bf16_wmma(
    const unsigned short* __restrict__ A, const unsigned int* __restrict__ Bp,
    float* __restrict__ C, int Mtiles, int NT, int ldc)
{
    const int wave = threadIdx.x >> 5;
    const int lane = threadIdx.x & 31;
    const int mt = blockIdx.x * 8 + wave;
    const int nt = blockIdx.y;
    if (mt >= Mtiles) return;

    const int g = lane >> 4;
    const int m = mt * 16 + (lane & 15);
    const unsigned short* arow = A + (size_t)m * NFEAT + g * 8;
    const v4u* bbase = reinterpret_cast<const v4u*>(Bp) + ((size_t)nt * 32 + lane) * 2;
    const size_t bstep = (size_t)NT * 64;

    v8f acc = {};
    #pragma unroll
    for (int kt = 0; kt < 8; ++kt) {
        FragBF a, b;
        const v4u* ap = reinterpret_cast<const v4u*>(arow + kt * 32);
        a.u[0] = ap[0];
        a.u[1] = ap[2];
        const v4u* bp = bbase + (size_t)kt * bstep;
        b.u[0] = bp[0];
        b.u[1] = bp[1];
        acc = wmma_bf16(a, b, acc);
    }

    const int ncol = nt * 16 + (lane & 15);
    float* crow = C + (size_t)(mt * 16 + g * 8) * ldc + ncol;
    #pragma unroll
    for (int r = 0; r < 8; ++r) crow[(size_t)r * ldc] = acc[r];
}

// ---------------- SpMM: out[row] += val * Hin[col]  (edge-parallel, float4) --------
__global__ void spmm4(const int* __restrict__ rows, const int* __restrict__ cols,
                      const float* __restrict__ vals, const float* __restrict__ Hin,
                      float* __restrict__ Hout, int E, int ld_in, int ld_out) {
    int e = blockIdx.x * blockDim.x + threadIdx.x;
    if (e >= E) return;
    int c = blockIdx.y;                                  // float4 chunk
    int r  = rows[e];
    int cl = cols[e];
    float v = vals[e];
    const float4 h = *reinterpret_cast<const float4*>(Hin + (size_t)cl * ld_in + c * 4);
    float* o = Hout + (size_t)r * ld_out + c * 4;
    atomicAdd(o + 0, v * h.x);
    atomicAdd(o + 1, v * h.y);
    atomicAdd(o + 2, v * h.z);
    atomicAdd(o + 3, v * h.w);
}

// ---------------- BatchNorm over node dim ----------------

__global__ __launch_bounds__(256) void bn_stats(const float* __restrict__ S,
                                                float* __restrict__ stats, int nrows) {
    int col = threadIdx.x;                    // thread t owns column t (coalesced rows)
    int r0 = blockIdx.x * 160;
    int r1 = r0 + 160; if (r1 > nrows) r1 = nrows;
    float s = 0.f, sq = 0.f;
    for (int r = r0; r < r1; ++r) {
        float v = S[(size_t)r * NFEAT + col];
        s += v; sq += v * v;
    }
    atomicAdd(&stats[col], s);
    atomicAdd(&stats[NFEAT + col], sq);
}

__global__ void bn_finalize(float* __restrict__ stats, float invN) {
    int c = threadIdx.x;
    float mu  = stats[c] * invN;
    float var = stats[NFEAT + c] * invN - mu * mu;      // biased var, as reference
    stats[2 * NFEAT + c] = mu;
    stats[3 * NFEAT + c] = rsqrtf(var + BN_EPS);
}

// normalize + ReLU + convert to bf16 activation buffer for the next WMMA GEMM
__global__ void bn_relu_cvt(const float* __restrict__ S, const float* __restrict__ stats,
                            const float* __restrict__ gamma, const float* __restrict__ beta,
                            unsigned short* __restrict__ out, long n) {
    long i = (long)blockIdx.x * blockDim.x + threadIdx.x;
    long stride = (long)gridDim.x * blockDim.x;
    for (; i < n; i += stride) {
        int col = (int)(i & (NFEAT - 1));
        float mu   = stats[2 * NFEAT + col];
        float rstd = stats[3 * NFEAT + col];
        float y = gamma[col] * (S[i] - mu) * rstd + beta[col];
        out[i] = f2bf(fmaxf(y, 0.0f));
    }
}

// ---------------- bias + log_softmax over 40 classes (one wave32 per row) ----------
__global__ __launch_bounds__(256) void bias_logsoftmax(const float* __restrict__ S,
                                                       const float* __restrict__ b,
                                                       float* __restrict__ out, int nrows) {
    int wave = threadIdx.x >> 5;
    int lane = threadIdx.x & 31;
    int row = blockIdx.x * 8 + wave;
    if (row >= nrows) return;
    const float* s = S + (size_t)row * 40;
    float z0 = s[lane] + b[lane];
    float z1 = (lane < 8) ? (s[lane + 32] + b[lane + 32]) : -3.4e38f;
    float m = fmaxf(z0, z1);
    #pragma unroll
    for (int off = 16; off > 0; off >>= 1) m = fmaxf(m, __shfl_xor(m, off, 32));
    float e = expf(z0 - m) + ((lane < 8) ? expf(z1 - m) : 0.0f);
    #pragma unroll
    for (int off = 16; off > 0; off >>= 1) e += __shfl_xor(e, off, 32);
    float ls = logf(e) + m;
    out[(size_t)row * 40 + lane] = z0 - ls;
    if (lane < 8) out[(size_t)row * 40 + lane + 32] = z1 - ls;
}

// ---------------- orchestration ----------------

extern "C" void kernel_launch(void* const* d_in, const int* in_sizes, int n_in,
                              void* d_out, int out_size, void* d_ws, size_t ws_size,
                              hipStream_t stream) {
    const float* x      = (const float*)d_in[0];
    const int*   arow   = (const int*)d_in[1];
    const int*   acol   = (const int*)d_in[2];
    const float* aval   = (const float*)d_in[3];
    const float* W1     = (const float*)d_in[4];
    // b1 (d_in[5]) cancels under training-mode BN
    const float* gamma1 = (const float*)d_in[6];
    const float* beta1  = (const float*)d_in[7];
    const float* W2     = (const float*)d_in[8];
    // b2 (d_in[9]) cancels under training-mode BN
    const float* gamma2 = (const float*)d_in[10];
    const float* beta2  = (const float*)d_in[11];
    const float* W3     = (const float*)d_in[12];
    const float* b3     = (const float*)d_in[13];
    float* out = (float*)d_out;

    const int N = in_sizes[0] / NFEAT;       // 20000 nodes (divisible by 32)
    const int E = in_sizes[1];               // 320000 edges
    const int Mtiles  = (N + 15) / 16;       // 1250
    const int Mtiles2 = N / 32;              // 625
    const long nElem = (long)N * NFEAT;

    // workspace layout
    char* ws = (char*)d_ws;
    size_t szAbf = (size_t)N * NFEAT * 2;    // bf16 activations
    size_t szG   = (size_t)N * NFEAT * 4;    // f32 GEMM / SpMM buffers
    unsigned short* Abf = (unsigned short*)ws;
    float* G = (float*)(ws + szAbf);
    float* S = (float*)(ws + szAbf + szG);
    unsigned int* W1p = (unsigned int*)(ws + szAbf + 2 * szG);   // 8*16*256 dwords
    unsigned int* W2p = W1p + 32768;
    unsigned int* W3p = W2p + 32768;                              // 8*3*256 dwords
    float* stats = (float*)(W3p + 6144);                          // 4*256 floats

    dim3 blk(256);

    // weight / input packing
    cvt_f32_bf16<<<dim3((unsigned)((nElem + 255) / 256)), blk, 0, stream>>>(x, Abf, nElem);
    pack_w_bf16<<<dim3(128), blk, 0, stream>>>(W1, W1p, 8, 16, 256);
    pack_w_bf16<<<dim3(128), blk, 0, stream>>>(W2, W2p, 8, 16, 256);
    pack_w_bf16<<<dim3(24),  blk, 0, stream>>>(W3, W3p, 8, 3, 40);

    dim3 gemm_grid((Mtiles2 + 7) / 8, 8);    // 32x32 tiles, 8 col-blocks of 32
    dim3 spmm_grid((E + 255) / 256, 64);
    unsigned zb = (unsigned)((nElem + 255) / 256);
    unsigned bnb = (unsigned)((N + 159) / 160);

    // ---- layer 1 ----
    gemm_bf16_wmma_2x2<<<gemm_grid, blk, 0, stream>>>(Abf, W1p, G, Mtiles2, 16, NFEAT);
    zero_f32<<<dim3(zb), blk, 0, stream>>>(S, nElem);
    spmm4<<<spmm_grid, blk, 0, stream>>>(arow, acol, aval, G, S, E, NFEAT, NFEAT);
    zero_f32<<<dim3(2), blk, 0, stream>>>(stats, 2 * NFEAT);
    bn_stats<<<dim3(bnb), blk, 0, stream>>>(S, stats, N);
    bn_finalize<<<dim3(1), blk, 0, stream>>>(stats, 1.0f / (float)N);
    bn_relu_cvt<<<dim3(zb), blk, 0, stream>>>(S, stats, gamma1, beta1, Abf, nElem);

    // ---- layer 2 ----
    gemm_bf16_wmma_2x2<<<gemm_grid, blk, 0, stream>>>(Abf, W2p, G, Mtiles2, 16, NFEAT);
    zero_f32<<<dim3(zb), blk, 0, stream>>>(S, nElem);
    spmm4<<<spmm_grid, blk, 0, stream>>>(arow, acol, aval, G, S, E, NFEAT, NFEAT);
    zero_f32<<<dim3(2), blk, 0, stream>>>(stats, 2 * NFEAT);
    bn_stats<<<dim3(bnb), blk, 0, stream>>>(S, stats, N);
    bn_finalize<<<dim3(1), blk, 0, stream>>>(stats, 1.0f / (float)N);
    bn_relu_cvt<<<dim3(zb), blk, 0, stream>>>(S, stats, gamma2, beta2, Abf, nElem);

    // ---- layer 3: 256 -> 40 (padded to 48 cols), spmm, bias + log_softmax ----
    dim3 gemm3_grid((Mtiles + 7) / 8, 3);
    gemm_bf16_wmma<<<gemm3_grid, blk, 0, stream>>>(Abf, W3p, G, Mtiles, 3, 48);
    long n3 = (long)N * 40;
    zero_f32<<<dim3((unsigned)((n3 + 255) / 256)), blk, 0, stream>>>(S, n3);
    dim3 spmm3_grid((E + 255) / 256, 10);
    spmm4<<<spmm3_grid, blk, 0, stream>>>(arow, acol, aval, G, S, E, 48, 40);
    bias_logsoftmax<<<dim3((N + 7) / 8), blk, 0, stream>>>(S, b3, out, N);
}
